// FCOS_70334384439406
// MI455X (gfx1250) — matrix operand order repaired and checked by reading, not compiled, not run on hardware
//
#include <hip/hip_runtime.h>
#include <hip/hip_bf16.h>

// ---------------------------------------------------------------------------
// FCOS head on gfx1250: bf16 WMMA implicit-GEMM convolutions with
// double-buffered async global->LDS staging (ASYNCcnt path) when available.
// ---------------------------------------------------------------------------

typedef __bf16 bf16_t;
typedef __attribute__((ext_vector_type(16))) bf16_t v16bf;
typedef __attribute__((ext_vector_type(8)))  float  v8f;
typedef int v4i_vec __attribute__((vector_size(16)));   // matches async builtin pointee

union Frag { uint4 u[2]; v16bf v; };

#if defined(__has_builtin)
#  if __has_builtin(__builtin_amdgcn_global_load_async_to_lds_b128) && \
      __has_builtin(__builtin_amdgcn_s_wait_asynccnt)
#    define USE_ASYNC 1
#  endif
#endif
#ifndef USE_ASYNC
#  define USE_ASYNC 0
#endif

#define AS1 __attribute__((address_space(1)))
#define AS3 __attribute__((address_space(3)))

__device__ __forceinline__ AS1 v4i_vec* to_as1(const void* p) {
    return (AS1 v4i_vec*)(unsigned long long)p;
}
__device__ __forceinline__ AS3 v4i_vec* to_as3(const void* p) {
    // generic LDS address carries the LDS offset in its low 32 bits
    return (AS3 v4i_vec*)(unsigned)(unsigned long long)p;
}

__device__ __forceinline__ unsigned bf16_rne(float f) {
    unsigned u = __float_as_uint(f);
    return (u + 0x7FFFu + ((u >> 16) & 1u)) >> 16;
}
__device__ __forceinline__ unsigned pack_bf16(float lo, float hi) {
    return bf16_rne(lo) | (bf16_rne(hi) << 16);
}

// ---------------------------------------------------------------------------
// Weight prep: fp32 [O][256][3][3] -> bf16-pair u32 [9][8][O][16]
// pair p of cblk cb covers input channels cb*32+2p, cb*32+2p+1.
// ---------------------------------------------------------------------------
__global__ void prep3x3(const float* __restrict__ w, unsigned* __restrict__ wA,
                        int O, int E) {
    for (int e = blockIdx.x * blockDim.x + threadIdx.x; e < E;
         e += gridDim.x * blockDim.x) {
        int p = e & 15;
        int r = e >> 4;
        int o = r % O; r /= O;
        int cb = r & 7;
        int t  = r >> 3;                    // tap = ky*3+kx
        int c = cb * 32 + 2 * p;
        float v0 = w[((size_t)o * 256 + c) * 9 + t];
        float v1 = w[((size_t)o * 256 + c + 1) * 9 + t];
        wA[e] = pack_bf16(v0, v1);
    }
}

// Combined head conv: rows 0-3 = bbox_w [4][256][3][3], row 4 = ctr_w [1][256][3][3]
__global__ void prep_head(const float* __restrict__ bw, const float* __restrict__ cw,
                          unsigned* __restrict__ wA) {
    const int E = 9 * 8 * 16 * 16;
    for (int e = blockIdx.x * blockDim.x + threadIdx.x; e < E;
         e += gridDim.x * blockDim.x) {
        int p = e & 15;
        int r = e >> 4;
        int o = r & 15; r >>= 4;
        int cb = r & 7;
        int t  = r >> 3;
        int c = cb * 32 + 2 * p;
        float v0 = 0.f, v1 = 0.f;
        if (o < 4) {
            v0 = bw[((size_t)o * 256 + c) * 9 + t];
            v1 = bw[((size_t)o * 256 + c + 1) * 9 + t];
        } else if (o == 4) {
            v0 = cw[(size_t)c * 9 + t];
            v1 = cw[((size_t)c + 1) * 9 + t];
        }
        wA[e] = pack_bf16(v0, v1);
    }
}

// 1x1 refine weights: fp32 [256][256] (o,c) -> u32 [1][8][256][16]
__global__ void prep1x1(const float* __restrict__ w, unsigned* __restrict__ wA) {
    const int E = 8 * 256 * 16;
    for (int e = blockIdx.x * blockDim.x + threadIdx.x; e < E;
         e += gridDim.x * blockDim.x) {
        int p = e & 15;
        int r = e >> 4;
        int o = r & 255;
        int cb = r >> 8;
        int c = cb * 32 + 2 * p;
        wA[e] = pack_bf16(w[(size_t)o * 256 + c], w[(size_t)o * 256 + c + 1]);
    }
}

__global__ void zero_fill(unsigned* __restrict__ p, int n) {
    int i = blockIdx.x * blockDim.x + threadIdx.x;
    if (i < n) p[i] = 0u;
}

// ---------------------------------------------------------------------------
// Activation swizzle: fp32 NCHW [2][256][S] -> bf16-pair u32 [2][8][S][16]
// ---------------------------------------------------------------------------
__global__ void swizzle_act(const float* __restrict__ in, unsigned* __restrict__ out,
                            int S, int E) {
    for (int e = blockIdx.x * blockDim.x + threadIdx.x; e < E;
         e += gridDim.x * blockDim.x) {
        int p = e & 15;
        int r = e >> 4;
        int n = r % S; r /= S;
        int cb = r & 7;
        int b  = r >> 3;
        int c = cb * 32 + 2 * p;
        float v0 = in[((size_t)b * 256 + c) * S + n];
        float v1 = in[((size_t)b * 256 + c + 1) * S + n];
        out[e] = pack_bf16(v0, v1);
    }
}

// ---------------------------------------------------------------------------
// WMMA implicit-GEMM conv. Block: 256 threads = 8 wave32, tile 128(M)x64(N).
// K pipeline: double-buffered 32ch x 64px bf16 LDS tiles; next tile staged
// (async when available) while WMMAs consume the current one.
// modes: 0 = fp32 NCHW out (+bias)           [logits / refine feats]
//        1 = relu(+bias), swizzled bf16 out   [tower intermediate]
//        2 = head: rows0-3 bbox relu(scale*(v+b)), row4 ctrness v+b2
// ---------------------------------------------------------------------------
struct ConvArgs {
    const unsigned* in_swz;   // [2][8][S][16] u32 (bf16 pairs)
    const unsigned* wA;       // [ntaps][8][Owt][16]
    const unsigned* zerobuf;  // 16 u32 of zeros (OOB async source)
    const float* bias;
    const float* bias2;       // ctr bias (mode 2)
    const float* scale_ptr;   // scales[5] (mode 2)
    float* out_f32;           // mode 0 main / mode 2 bbox
    float* out_ctr;           // mode 2 ctrness
    unsigned* out_swz;        // mode 1
    int S, H, W, logW;
    int Owt, Ologic, ntaps, mode, level;
};

#define LDS_STRIDE 20   // u32 stride per pixel row (16 pairs + pad), 16B aligned

__global__ __launch_bounds__(256, 2) void conv_wmma_bf16(ConvArgs a) {
    __shared__ unsigned Blds[2][64 * LDS_STRIDE];

    const int tid  = threadIdx.x;
    const int lane = tid & 31;
    const int wv   = tid >> 5;
    const int b    = blockIdx.z;
    const int n0   = blockIdx.x * 64;
    const int o0   = blockIdx.y * 128 + wv * 16;
    const bool active = (o0 < a.Owt);

    const int m = lane & 15;      // A row within tile / B column within subtile
    const int g = lane >> 4;      // lane-group

    const unsigned* inb = a.in_swz + (size_t)b * 8 * a.S * 16;

    v8f acc[4];
#pragma unroll
    for (int i = 0; i < 4; ++i) acc[i] = (v8f)0.0f;

    // staging coords for this thread: one 16B chunk per K-step
    const int sn  = tid >> 2;   // pixel 0..63
    const int spq = tid & 3;    // pair-quad 0..3
    const int np_st = n0 + sn;
    const int sy = np_st >> a.logW;
    const int sx = np_st & (a.W - 1);
    const bool is3 = (a.ntaps == 9);
    const int nsteps = a.ntaps * 8;

    // stage K-step `step` into buffer `buf`
    auto issue = [&](int step, unsigned* buf) {
        const int t  = step >> 3;
        const int cb = step & 7;
        const int dy = is3 ? (t / 3 - 1) : 0;
        const int dx = is3 ? (t % 3 - 1) : 0;
        const int yy = sy + dy;
        const int xx = sx + dx;
        const bool ok = ((unsigned)yy < (unsigned)a.H) &&
                        ((unsigned)xx < (unsigned)a.W);
        const int nsrc = (yy << a.logW) | (xx & (a.W - 1));
        const unsigned* src = ok
            ? (inb + ((size_t)cb * a.S + nsrc) * 16 + spq * 4)
            : a.zerobuf;
        unsigned* dst = &buf[sn * LDS_STRIDE + spq * 4];
#if USE_ASYNC
        __builtin_amdgcn_global_load_async_to_lds_b128(to_as1(src), to_as3(dst), 0, 0);
#else
        *(uint4*)dst = *(const uint4*)src;
#endif
    };

    // prime the pipeline
    issue(0, Blds[0]);
#if USE_ASYNC
    __builtin_amdgcn_s_wait_asynccnt(0);
#endif
    __syncthreads();

    for (int step = 0; step < nsteps; ++step) {
        unsigned* cur = Blds[step & 1];
        if (step + 1 < nsteps) issue(step + 1, Blds[(step + 1) & 1]);

        if (active) {
            const int t  = step >> 3;
            const int cb = step & 7;
            // ---- A fragment: weights, two b128 per lane ----
            const unsigned* wbase = a.wA +
                (((size_t)t * 8 + cb) * a.Owt + (o0 + m)) * 16;
            __builtin_prefetch(wbase + (size_t)8 * a.Owt * 16, 0, 1);
            Frag af;
            af.u[0] = *(const uint4*)(wbase + 4 * g);
            af.u[1] = *(const uint4*)(wbase + 8 + 4 * g);
#pragma unroll
            for (int s2 = 0; s2 < 4; ++s2) {
                // ---- B fragment: pairs {4g..4g+3, 8+4g..8+4g+3} at pixel column ----
                const int n = s2 * 16 + m;
                Frag bf;
                bf.u[0] = *(const uint4*)(&cur[n * LDS_STRIDE + 4 * g]);
                bf.u[1] = *(const uint4*)(&cur[n * LDS_STRIDE + 8 + 4 * g]);
                acc[s2] = __builtin_amdgcn_wmma_f32_16x16x32_bf16(
                    false, af.v, false, bf.v, (short)0, acc[s2],
                    false, false);
            }
        }
#if USE_ASYNC
        __builtin_amdgcn_s_wait_asynccnt(0);
#endif
        __syncthreads();
    }

    if (!active) return;

    // ------------------------------ epilogue -------------------------------
    if (a.mode == 1) {
        // relu(+bias) -> swizzled bf16
        unsigned* osw = a.out_swz + (size_t)b * 8 * a.S * 16;
        const int ob    = o0 + 8 * g;          // first of this lane's 8 channels
        const int cbO   = ob >> 5;
        const int pbase = (ob & 31) >> 1;
#pragma unroll
        for (int s2 = 0; s2 < 4; ++s2) {
            const int np = n0 + s2 * 16 + m;
            unsigned pw[4];
#pragma unroll
            for (int q = 0; q < 4; ++q) {
                float v0 = fmaxf(acc[s2][2 * q]     + a.bias[ob + 2 * q],     0.f);
                float v1 = fmaxf(acc[s2][2 * q + 1] + a.bias[ob + 2 * q + 1], 0.f);
                pw[q] = pack_bf16(v0, v1);
            }
            *(uint4*)(osw + ((size_t)cbO * a.S + np) * 16 + pbase) =
                make_uint4(pw[0], pw[1], pw[2], pw[3]);
        }
    } else if (a.mode == 0) {
        float* out = a.out_f32 + (size_t)b * a.Ologic * a.S;
#pragma unroll
        for (int s2 = 0; s2 < 4; ++s2) {
            const int np = n0 + s2 * 16 + m;
#pragma unroll
            for (int r = 0; r < 8; ++r) {
                const int o = o0 + r + 8 * g;
                if (o < a.Ologic)
                    out[(size_t)o * a.S + np] = acc[s2][r] + a.bias[o];
            }
        }
    } else {
        // head: o<4 -> bbox relu(scale*(v+b)), o==4 -> ctrness
        const float sc = a.scale_ptr[a.level];
        float* bb = a.out_f32 + (size_t)b * 4 * a.S;
        float* ct = a.out_ctr + (size_t)b * a.S;
#pragma unroll
        for (int s2 = 0; s2 < 4; ++s2) {
            const int np = n0 + s2 * 16 + m;
#pragma unroll
            for (int r = 0; r < 8; ++r) {
                const int o = o0 + r + 8 * g;
                if (o < 4)
                    bb[(size_t)o * a.S + np] =
                        fmaxf(sc * (acc[s2][r] + a.bias[o]), 0.f);
                else if (o == 4)
                    ct[np] = acc[s2][r] + a.bias2[0];
            }
        }
    }
}

// ---------------------------------------------------------------------------
// Host orchestration
// ---------------------------------------------------------------------------
extern "C" void kernel_launch(void* const* d_in, const int* in_sizes, int n_in,
                              void* d_out, int out_size, void* d_ws, size_t ws_size,
                              hipStream_t stream) {
    (void)in_sizes; (void)n_in; (void)out_size; (void)ws_size;
    static const int S_[5]  = {16384, 4096, 1024, 256, 64};
    static const int HW_[5] = {128, 64, 32, 16, 8};
    static const int LW_[5] = {7, 6, 5, 4, 3};

    const float* srm[5];  const float* ksrm[5];
    for (int l = 0; l < 5; ++l) {
        srm[l]  = (const float*)d_in[10 + l];
        ksrm[l] = (const float*)d_in[15 + l];
    }
    const float* cls_tw  = (const float*)d_in[20];
    const float* cls_tb  = (const float*)d_in[21];
    const float* box_tw  = (const float*)d_in[22];
    const float* box_tb  = (const float*)d_in[23];
    const float* logit_w = (const float*)d_in[24];
    const float* logit_b = (const float*)d_in[25];
    const float* bbox_w  = (const float*)d_in[26];
    const float* bbox_b  = (const float*)d_in[27];
    const float* ctr_w   = (const float*)d_in[28];
    const float* ctr_b   = (const float*)d_in[29];
    const float* scales  = (const float*)d_in[30];
    const float* ref_w   = (const float*)d_in[31];
    const float* ref_b   = (const float*)d_in[32];
    const float* refk_w  = (const float*)d_in[33];
    const float* refk_b  = (const float*)d_in[34];

    // ---- workspace carve-up (u32 units, all offsets multiple of 16) ----
    unsigned* ws = (unsigned*)d_ws;
    size_t off = 0;
    auto take = [&](size_t n) { unsigned* p = ws + off; off += n; return p; };
    unsigned* zerobuf = take(16);
    unsigned* wA_cls[4]; for (int i = 0; i < 4; ++i) wA_cls[i] = take(9 * 8 * 256 * 16);
    unsigned* wA_box[4]; for (int i = 0; i < 4; ++i) wA_box[i] = take(9 * 8 * 256 * 16);
    unsigned* wA_logit = take(9 * 8 * 80 * 16);
    unsigned* wA_head  = take(9 * 8 * 16 * 16);
    unsigned* wA_ref[5];  for (int l = 0; l < 5; ++l) wA_ref[l]  = take(8 * 256 * 16);
    unsigned* wA_refk[5]; for (int l = 0; l < 5; ++l) wA_refk[l] = take(8 * 256 * 16);
    unsigned* swzA = take((size_t)2 * 8 * 16384 * 16);
    unsigned* swzK = take((size_t)2 * 8 * 16384 * 16);
    unsigned* ping = take((size_t)2 * 8 * 16384 * 16);
    unsigned* pong = take((size_t)2 * 8 * 16384 * 16);

    // ---- output offsets (return order: logits*5, bbox*5, ctr*5, feats*5, feats_k*5)
    float* outF = (float*)d_out;
    size_t o_log[5], o_bb[5], o_ct[5], o_ft[5], o_fk[5], cur = 0;
    for (int l = 0; l < 5; ++l) { o_log[l] = cur; cur += (size_t)2 * 80 * S_[l]; }
    for (int l = 0; l < 5; ++l) { o_bb[l]  = cur; cur += (size_t)2 * 4  * S_[l]; }
    for (int l = 0; l < 5; ++l) { o_ct[l]  = cur; cur += (size_t)2 * 1  * S_[l]; }
    for (int l = 0; l < 5; ++l) { o_ft[l]  = cur; cur += (size_t)2 * 256 * S_[l]; }
    for (int l = 0; l < 5; ++l) { o_fk[l]  = cur; cur += (size_t)2 * 256 * S_[l]; }

    auto gsz = [](int E) { int b = (E + 255) / 256; return b > 4096 ? 4096 : b; };

    zero_fill<<<1, 16, 0, stream>>>(zerobuf, 16);

    // ---- weight prep (level-independent) ----
    const int E256 = 9 * 8 * 256 * 16;
    for (int i = 0; i < 4; ++i) {
        prep3x3<<<gsz(E256), 256, 0, stream>>>(cls_tw + (size_t)i * 256 * 256 * 9,
                                               wA_cls[i], 256, E256);
        prep3x3<<<gsz(E256), 256, 0, stream>>>(box_tw + (size_t)i * 256 * 256 * 9,
                                               wA_box[i], 256, E256);
    }
    const int E80 = 9 * 8 * 80 * 16;
    prep3x3<<<gsz(E80), 256, 0, stream>>>(logit_w, wA_logit, 80, E80);
    prep_head<<<gsz(9 * 8 * 16 * 16), 256, 0, stream>>>(bbox_w, ctr_w, wA_head);
    for (int l = 0; l < 5; ++l) {
        prep1x1<<<gsz(8 * 256 * 16), 256, 0, stream>>>(ref_w  + (size_t)l * 65536, wA_ref[l]);
        prep1x1<<<gsz(8 * 256 * 16), 256, 0, stream>>>(refk_w + (size_t)l * 65536, wA_refk[l]);
    }

    auto conv = [&](const unsigned* in_swz, const unsigned* wA, const float* bias,
                    int l, int Owt, int Ologic, int ntaps, int mode,
                    float* out_f32, float* out_ctr, unsigned* out_swz,
                    const float* bias2) {
        ConvArgs A;
        A.in_swz = in_swz; A.wA = wA; A.zerobuf = zerobuf;
        A.bias = bias; A.bias2 = bias2; A.scale_ptr = scales;
        A.out_f32 = out_f32; A.out_ctr = out_ctr; A.out_swz = out_swz;
        A.S = S_[l]; A.H = HW_[l]; A.W = HW_[l]; A.logW = LW_[l];
        A.Owt = Owt; A.Ologic = Ologic; A.ntaps = ntaps; A.mode = mode; A.level = l;
        dim3 g(S_[l] / 64, (Owt + 127) / 128, 2);
        conv_wmma_bf16<<<g, 256, 0, stream>>>(A);
    };

    for (int l = 0; l < 5; ++l) {
        const int S = S_[l];
        const int Esw = 2 * 8 * S * 16;
        swizzle_act<<<gsz(Esw), 256, 0, stream>>>(srm[l],  swzA, S, Esw);
        swizzle_act<<<gsz(Esw), 256, 0, stream>>>(ksrm[l], swzK, S, Esw);

        // refine / refine_k 1x1 convs -> feats, feats_k (fp32 NCHW)
        conv(swzA, wA_ref[l],  ref_b  + (size_t)l * 256, l, 256, 256, 1, 0,
             outF + o_ft[l], nullptr, nullptr, nullptr);
        conv(swzK, wA_refk[l], refk_b + (size_t)l * 256, l, 256, 256, 1, 0,
             outF + o_fk[l], nullptr, nullptr, nullptr);

        // cls tower (4x conv3 + relu), then logits head
        conv(swzA, wA_cls[0], cls_tb + 0 * 256, l, 256, 256, 9, 1, nullptr, nullptr, ping, nullptr);
        conv(ping, wA_cls[1], cls_tb + 1 * 256, l, 256, 256, 9, 1, nullptr, nullptr, pong, nullptr);
        conv(pong, wA_cls[2], cls_tb + 2 * 256, l, 256, 256, 9, 1, nullptr, nullptr, ping, nullptr);
        conv(ping, wA_cls[3], cls_tb + 3 * 256, l, 256, 256, 9, 1, nullptr, nullptr, pong, nullptr);
        conv(pong, wA_logit, logit_b, l, 80, 80, 9, 0,
             outF + o_log[l], nullptr, nullptr, nullptr);

        // box tower (4x conv3 + relu), then combined bbox+ctr head
        conv(swzA, wA_box[0], box_tb + 0 * 256, l, 256, 256, 9, 1, nullptr, nullptr, ping, nullptr);
        conv(ping, wA_box[1], box_tb + 1 * 256, l, 256, 256, 9, 1, nullptr, nullptr, pong, nullptr);
        conv(pong, wA_box[2], box_tb + 2 * 256, l, 256, 256, 9, 1, nullptr, nullptr, ping, nullptr);
        conv(ping, wA_box[3], box_tb + 3 * 256, l, 256, 256, 9, 1, nullptr, nullptr, pong, nullptr);
        conv(pong, wA_head, bbox_b, l, 16, 16, 9, 2,
             outF + o_bb[l], outF + o_ct[l], nullptr, ctr_b);
    }
}
